// IGMC_82437602279937
// MI455X (gfx1250) — compile-verified, matrix operand order checked by reading, not run on hardware
//
#include <hip/hip_runtime.h>
#include <math.h>

// ---------------------------------------------------------------------------
// IGMC pipeline for MI455X (gfx1250, wave32).
//   RGCN layer  = scatter-aggregate per (dst,relation)  ->  one f16 WMMA GEMM
//                 [N, 6*32] x [6*32, 32]  (slot 5 = self/root, pre-scaled 1/deg)
//   Operands pre-swizzled into wave32 WMMA fragment order so the GEMM inner
//   loop is pure {global_load_b128, ds-resident B in VGPRs, v_wmma}.
//   Walk        = one workgroup per graph, sequential 21-step greedy walk in LDS
//   Head        = 128x128x128 WMMA GEMM + wave32 shuffle-reduced log-softmax
// ---------------------------------------------------------------------------

#define BGR   128      // graphs
#define NPER  512      // nodes per graph
#define NREL  5
#define NBAS  2
#define MAXN  50
#define WALKS 21
#define NC    5
#define DIMD  128
#define KCAT  192      // 6 slots * 32
#define KSTEPS 6
#define TPW   4        // M-tiles per wave in the RGCN GEMM

typedef _Float16 v8h  __attribute__((ext_vector_type(8)));
typedef _Float16 v16h __attribute__((ext_vector_type(16)));
typedef float    v8f  __attribute__((ext_vector_type(8)));

// ---------------- setup ----------------------------------------------------

__global__ void k_zero_degcur(float* __restrict__ deg, int* __restrict__ cur, int n) {
    int i = blockIdx.x * blockDim.x + threadIdx.x;
    if (i < n) { deg[i] = 0.0f; cur[i] = 0; }
}

// in-degree (float, for mean agg), out-degree + unsorted neighbor table
__global__ void k_deg_table(const int* __restrict__ src, const int* __restrict__ dst,
                            float* __restrict__ deg, int* __restrict__ cur,
                            int* __restrict__ table, int E) {
    int e = blockIdx.x * blockDim.x + threadIdx.x;
    if (e >= E) return;
    int s = src[e], d = dst[e];
    atomicAdd(&deg[d], 1.0f);
    int rk = atomicAdd(&cur[s], 1);
    if (rk < MAXN) table[(size_t)s * MAXN + rk] = d;
}

// sort each node's neighbor list ascending (matches lexsort by (src,dst))
__global__ void k_sort_table(int* __restrict__ table, const int* __restrict__ cur, int n) {
    int v = blockIdx.x * blockDim.x + threadIdx.x;
    if (v >= n) return;
    int m = cur[v]; if (m > MAXN) m = MAXN;
    int* t = table + (size_t)v * MAXN;
    for (int i = 1; i < m; ++i) {
        int key = t[i], j = i - 1;
        while (j >= 0 && t[j] > key) { t[j + 1] = t[j]; --j; }
        t[j + 1] = key;
    }
}

// ---------------- per-layer RGCN -------------------------------------------

// logical weight element W[kk][col], kk in [0,192): rows r*32+k are
// sum_b comp[r,b]*basis[b,k,:] for r<5 (k<fin, else 0), rows 160+k are root.
__device__ __forceinline__ float wcat_elem(const float* basis, const float* comp,
                                           const float* root, int fin, int kk, int col) {
    int r = kk >> 5, k = kk & 31;
    float v = 0.0f;
    if (k < fin) {
        if (r < NREL) {
            for (int b = 0; b < NBAS; ++b)
                v += comp[r * NBAS + b] * basis[((size_t)b * fin + k) * 32 + col];
        } else {
            v = root[(size_t)k * 32 + col];
        }
    }
    return v;
}

// Pack weights directly in per-lane B-fragment order:
//   wpack[(s*2+nt)*512 + lane*16 + i]  (i<8 -> K=kb+koff+i, i>=8 -> K=kb+16+koff+i-8)
__global__ void k_build_wcat(const float* __restrict__ basis, const float* __restrict__ comp,
                             const float* __restrict__ root, _Float16* __restrict__ wpack,
                             int fin) {
    int idx = blockIdx.x * blockDim.x + threadIdx.x;
    if (idx >= KSTEPS * 2 * 32 * 16) return;          // 6144
    int j    = idx >> 9;                               // s*2+nt
    int lane = (idx >> 4) & 31;
    int i    = idx & 15;
    int s = j >> 1, nt = j & 1;
    int koff = (lane >> 4) << 3;
    int col  = (lane & 15) + nt * 16;
    int kk   = s * 32 + ((i < 8) ? (koff + i) : (16 + koff + (i - 8)));
    wpack[idx] = (_Float16)wcat_elem(basis, comp, root, fin, kk, col);
}

// agg[N][6][32]: zero relation slots, slot 5 = h[n]
__global__ void k_agg_init(const float* __restrict__ h, int hstride, int fin,
                           float* __restrict__ agg, unsigned total) {
    unsigned idx = blockIdx.x * blockDim.x + threadIdx.x;
    if (idx >= total) return;
    unsigned n = idx / KCAT, c = idx % KCAT;
    float v = 0.0f;
    if (c >= 160u) { int k = (int)c - 160; if (k < fin) v = h[(size_t)n * hstride + k]; }
    agg[idx] = v;
}

// agg[dst][etype][k] += h[src][k]
__global__ void k_agg_scatter(const int* __restrict__ src, const int* __restrict__ dst,
                              const int* __restrict__ et, const float* __restrict__ h,
                              int hstride, int fsh, int fin,
                              float* __restrict__ agg, unsigned total) {
    unsigned idx = blockIdx.x * blockDim.x + threadIdx.x;
    if (idx >= total) return;
    unsigned e = idx >> fsh;
    int k = (int)(idx & (unsigned)(fin - 1));
    int s = src[e], d = dst[e], r = et[e];
    const float* hp = h + (size_t)s * hstride;
    __builtin_prefetch(hp, 0, 0);                    // global_prefetch_b8
    atomicAdd(&agg[(size_t)d * KCAT + (size_t)r * 32 + k], hp[k]);
}

// scale relation slots by 1/max(deg,1), convert to f16, and store in swizzled
// A-fragment order: aggswz[((tile*6+s)*32 + lane)*16 + i]
__global__ void k_scale_half(const float* __restrict__ agg, const float* __restrict__ deg,
                             _Float16* __restrict__ aggswz, unsigned total) {
    unsigned idx = blockIdx.x * blockDim.x + threadIdx.x;
    if (idx >= total) return;
    unsigned n = idx / KCAT, c = idx % KCAT;
    float dg = deg[n]; if (dg < 1.0f) dg = 1.0f;
    float sc = ((c >> 5) < NREL) ? (1.0f / dg) : 1.0f;
    float v = agg[idx] * sc;
    unsigned t = n >> 4, m = n & 15;
    unsigned s = c >> 5, kis = c & 31;
    unsigned grp = (kis >> 3) & 1u;                   // lane group (K half)
    unsigned i   = (kis & 7u) | ((kis >> 4) << 3);    // fragment element
    unsigned lane = (grp << 4) | m;
    aggswz[(((size_t)t * KSTEPS + s) * 32 + lane) * 16 + i] = (_Float16)v;
}

// D = tanh( A[N,192] @ W[192,32] + bias )  -> cs[:, l*32 : l*32+32]
// B fragments live in VGPRs across TPW M-tiles; A is one 32B load per K-step.
__global__ __launch_bounds__(128)
void k_rgcn_gemm(const _Float16* __restrict__ aggswz, const _Float16* __restrict__ wpack,
                 const float* __restrict__ cbias, float* __restrict__ cs,
                 int layer, int nTiles) {
    __shared__ _Float16 wsh[KSTEPS * 2 * 32 * 16];    // 12 KB
    int tid = threadIdx.x;
    const int4* wsrcv = (const int4*)wpack;
    int4* wdstv = (int4*)wsh;
    for (int i = tid; i < (KSTEPS * 2 * 32 * 16) / 8; i += 128) wdstv[i] = wsrcv[i];
    __syncthreads();

    int wave = tid >> 5, lane = tid & 31;
    v16h bf[KSTEPS * 2];
#pragma unroll
    for (int j = 0; j < KSTEPS * 2; ++j)
        bf[j] = *(const v16h*)(wsh + ((size_t)(j * 32 + lane)) * 16);

    int base = (blockIdx.x * 4 + wave) * TPW;
    int colb = lane & 15;
    int roff = (lane >> 4) << 3;
    for (int t = 0; t < TPW; ++t) {
        int mtile = base + t;
        if (mtile >= nTiles) break;                   // wave-uniform
        const _Float16* ap = aggswz + ((size_t)mtile * KSTEPS * 32 + lane) * 16;
        v8f acc0 = {}; v8f acc1 = {};
#pragma unroll
        for (int s = 0; s < KSTEPS; ++s) {
            v16h a = *(const v16h*)(ap + (size_t)s * 32 * 16);
            acc0 = __builtin_amdgcn_wmma_f32_16x16x32_f16(false, a, false, bf[s * 2],     (short)0, acc0, false, false);
            acc1 = __builtin_amdgcn_wmma_f32_16x16x32_f16(false, a, false, bf[s * 2 + 1], (short)0, acc1, false, false);
        }
        int rbase = mtile * 16 + roff;
#pragma unroll
        for (int i = 0; i < 8; ++i) {
            float v0 = tanhf(acc0[i] + cbias[colb]);
            float v1 = tanhf(acc1[i] + cbias[colb + 16]);
            cs[(size_t)(rbase + i) * DIMD + layer * 32 + colb]      = v0;
            cs[(size_t)(rbase + i) * DIMD + layer * 32 + colb + 16] = v1;
        }
    }
}

// ---------------- greedy reasoning walk ------------------------------------

__global__ __launch_bounds__(128)
void k_walk(const float* __restrict__ cs, const int* __restrict__ table,
            const int* __restrict__ degout, const float* __restrict__ x,
            const float* __restrict__ walk_w, const float* __restrict__ walk_b,
            float* __restrict__ pooled) {
    int g = blockIdx.x, tid = threadIdx.x;
    __shared__ float acc[DIMD];
    __shared__ unsigned trav[NPER / 32];
    __shared__ float s_logit[MAXN];
    __shared__ int   s_nb[MAXN];
    __shared__ int s_node, s_done, s_add, s_start, s_end, s_count;

    if (tid == 0) {
        int st = 0, en = 0; float bs = -3.0e38f, be = -3.0e38f;
        for (int n = 0; n < NPER; ++n) {
            float v0 = x[(size_t)(g * NPER + n) * 4 + 0];
            float v1 = x[(size_t)(g * NPER + n) * 4 + 1];
            if (v0 > bs) { bs = v0; st = n; }
            if (v1 > be) { be = v1; en = n; }
        }
        s_start = st; s_end = en; s_node = st; s_done = 0;
    }
    if (tid < NPER / 32) trav[tid] = 0u;
    __syncthreads();
    if (tid == 0) trav[s_start >> 5] |= 1u << (s_start & 31);
    const float* xg = cs + (size_t)g * NPER * DIMD;
    acc[tid] = xg[(size_t)s_node * DIMD + tid];
    __syncthreads();

    for (int step = 0; step < WALKS; ++step) {
        int cur = s_node;
        int row = g * NPER + cur;
        int cnt = degout[row];
        if (tid < MAXN) {
            int nbg = table[(size_t)row * MAXN + tid];
            int nbl = nbg - g * NPER;
            if (nbl < 0) nbl = 0; if (nbl > NPER - 1) nbl = NPER - 1;
            float lg = walk_b[tid];
            for (int k = 0; k < DIMD; ++k) lg += walk_w[tid * DIMD + k] * acc[k];
            bool visited = (trav[nbl >> 5] >> (nbl & 31)) & 1u;
            bool valid = (tid < cnt) && !visited;
            s_nb[tid] = nbl;
            s_logit[tid] = valid ? lg : -3.0e38f;
        }
        __syncthreads();
        if (tid == 0) {
            int best = 0; float bv = s_logit[0];
            for (int j = 1; j < MAXN; ++j)
                if (s_logit[j] > bv) { bv = s_logit[j]; best = j; }
            int sel = s_nb[best];
            int no_nbr = (cnt == 0);
            int addidx = -1;
            if (!s_done) {
                addidx = no_nbr ? s_end : sel;
                if (!no_nbr) { trav[sel >> 5] |= 1u << (sel & 31); s_node = sel; }
                if (no_nbr || sel == s_end) s_done = 1;
            }
            s_add = addidx;
        }
        __syncthreads();
        if (s_add >= 0) acc[tid] += xg[(size_t)s_add * DIMD + tid];
        __syncthreads();
    }
    if (tid == 0) {
        int c = 0;
        for (int i = 0; i < NPER / 32; ++i) c += __popc(trav[i]);
        s_count = c;
    }
    __syncthreads();
    pooled[(size_t)g * DIMD + tid] = acc[tid] / (float)s_count;
}

// ---------------- MLP head --------------------------------------------------

// hid = relu( pooled[128,128] @ lin1_w^T + b1 ), one 16x16 WMMA tile per wave
__global__ __launch_bounds__(32)
void k_head(const float* __restrict__ pooled, const float* __restrict__ w1,
            const float* __restrict__ b1, float* __restrict__ hid) {
    int lane = threadIdx.x;
    int mt = blockIdx.x >> 3, nt = blockIdx.x & 7;
    int mm = lane & 15, koff = (lane >> 4) << 3;
    const float* ar = pooled + (size_t)(mt * 16 + mm) * DIMD;
    const float* br = w1 + (size_t)(nt * 16 + mm) * DIMD;   // B[k][n] = w1[n][k]
    v8f acc = {};
#pragma unroll
    for (int s = 0; s < 4; ++s) {
        int kb = s * 32;
        v16h a, b;
#pragma unroll
        for (int i = 0; i < 8; ++i) {
            a[i]     = (_Float16)ar[kb + koff + i];
            a[i + 8] = (_Float16)ar[kb + 16 + koff + i];
            b[i]     = (_Float16)br[kb + koff + i];
            b[i + 8] = (_Float16)br[kb + 16 + koff + i];
        }
        acc = __builtin_amdgcn_wmma_f32_16x16x32_f16(false, a, false, b, (short)0, acc, false, false);
    }
    int rbase = mt * 16 + ((lane >> 4) << 3);
    int col = nt * 16 + (lane & 15);
#pragma unroll
    for (int i = 0; i < 8; ++i) {
        float v = acc[i] + b1[col];
        hid[(size_t)(rbase + i) * DIMD + col] = v > 0.0f ? v : 0.0f;
    }
}

// logits + log_softmax; one wave32 per graph, butterfly shuffle reduction
__global__ __launch_bounds__(32)
void k_logits(const float* __restrict__ hid, const float* __restrict__ w2,
              const float* __restrict__ b2, float* __restrict__ out) {
    int g = blockIdx.x, lane = threadIdx.x;
    float lg[NC];
#pragma unroll
    for (int c = 0; c < NC; ++c) {
        float p = 0.0f;
        for (int k = lane; k < DIMD; k += 32)
            p += hid[(size_t)g * DIMD + k] * w2[(size_t)c * DIMD + k];
#pragma unroll
        for (int off = 16; off > 0; off >>= 1) p += __shfl_xor(p, off, 32);
        lg[c] = p + b2[c];
    }
    float mx = lg[0];
#pragma unroll
    for (int c = 1; c < NC; ++c) mx = lg[c] > mx ? lg[c] : mx;
    float se = 0.0f;
#pragma unroll
    for (int c = 0; c < NC; ++c) se += expf(lg[c] - mx);
    float lse = mx + logf(se);
    if (lane < NC) {
        float myl = 0.0f;
#pragma unroll
        for (int c = 0; c < NC; ++c) if (c == lane) myl = lg[c];
        out[(size_t)g * NC + lane] = myl - lse;
    }
}

// ---------------- launch ----------------------------------------------------

extern "C" void kernel_launch(void* const* d_in, const int* in_sizes, int n_in,
                              void* d_out, int out_size, void* d_ws, size_t ws_size,
                              hipStream_t stream) {
    const float* x  = (const float*)d_in[0];
    const int*   ei = (const int*)d_in[1];
    const int*   et = (const int*)d_in[2];
    const int E = in_sizes[2];
    const int N = in_sizes[3];
    const int* src = ei;
    const int* dst = ei + E;
    const float *basis[4], *comp[4], *root[4], *cbias[4];
    for (int l = 0; l < 4; ++l) {
        basis[l] = (const float*)d_in[4 + l * 4 + 0];
        comp[l]  = (const float*)d_in[4 + l * 4 + 1];
        root[l]  = (const float*)d_in[4 + l * 4 + 2];
        cbias[l] = (const float*)d_in[4 + l * 4 + 3];
    }
    const float* walk_w = (const float*)d_in[20];
    const float* walk_b = (const float*)d_in[21];
    const float* lin1w  = (const float*)d_in[22];
    const float* lin1b  = (const float*)d_in[23];
    const float* lin2w  = (const float*)d_in[24];
    const float* lin2b  = (const float*)d_in[25];
    float* out = (float*)d_out;

    // workspace carve-up (whole footprint ~120 MB -> resident in 192 MB L2)
    char* ws = (char*)d_ws;
    size_t off = 0;
    auto carve = [&](size_t bytes) -> void* {
        void* p = (void*)(ws + off);
        off = (off + bytes + 255) & ~(size_t)255;
        return p;
    };
    float*    deg    = (float*)   carve((size_t)N * 4);
    int*      cursor = (int*)     carve((size_t)N * 4);
    int*      table  = (int*)     carve((size_t)N * MAXN * 4);
    float*    agg    = (float*)   carve((size_t)N * KCAT * 4);
    _Float16* aggswz = (_Float16*)carve((size_t)N * KCAT * 2);
    _Float16* wpack  = (_Float16*)carve((size_t)KSTEPS * 2 * 32 * 16 * 2);
    float*    cs     = (float*)   carve((size_t)N * DIMD * 4);
    float*    pooled = (float*)   carve((size_t)BGR * DIMD * 4);
    float*    hid    = (float*)   carve((size_t)BGR * DIMD * 4);

    k_zero_degcur<<<(N + 255) / 256, 256, 0, stream>>>(deg, cursor, N);
    k_deg_table<<<(E + 255) / 256, 256, 0, stream>>>(src, dst, deg, cursor, table, E);
    k_sort_table<<<(N + 255) / 256, 256, 0, stream>>>(table, cursor, N);

    const unsigned nkc = (unsigned)N * KCAT;
    const int nTiles = N / 16;
    const int gemmBlocks = (nTiles + 4 * TPW - 1) / (4 * TPW);
    for (int l = 0; l < 4; ++l) {
        const int fin = (l == 0) ? 4 : 32;
        const int fsh = (l == 0) ? 2 : 5;
        const int hstride = (l == 0) ? 4 : DIMD;
        const float* h = (l == 0) ? x : (const float*)(cs + (size_t)(l - 1) * 32);
        const unsigned esc = (unsigned)E << fsh;

        k_build_wcat<<<(KSTEPS * 2 * 32 * 16 + 255) / 256, 256, 0, stream>>>(basis[l], comp[l], root[l], wpack, fin);
        k_agg_init<<<(nkc + 255u) / 256u, 256, 0, stream>>>(h, hstride, fin, agg, nkc);
        k_agg_scatter<<<(esc + 255u) / 256u, 256, 0, stream>>>(src, dst, et, h, hstride, fsh, fin, agg, esc);
        k_scale_half<<<(nkc + 255u) / 256u, 256, 0, stream>>>(agg, deg, aggswz, nkc);
        k_rgcn_gemm<<<gemmBlocks, 128, 0, stream>>>(aggswz, wpack, cbias[l], cs, l, nTiles);
    }

    k_walk<<<BGR, 128, 0, stream>>>(cs, table, cursor, x, walk_w, walk_b, pooled);
    k_head<<<64, 32, 0, stream>>>(pooled, lin1w, lin1b, hid);
    k_logits<<<BGR, 32, 0, stream>>>(hid, lin2w, lin2b, out);
}
